// SelectiveSSM_43696997269907
// MI455X (gfx1250) — matrix operand order repaired and checked by reading, not compiled, not run on hardware
//
#include <hip/hip_runtime.h>
#include <hip/hip_bf16.h>

// ---------------------------------------------------------------------------
// Types / helpers
// ---------------------------------------------------------------------------
typedef __attribute__((ext_vector_type(16))) __bf16        v16bf;
typedef __attribute__((ext_vector_type(8)))  float         v8f;
typedef __attribute__((ext_vector_type(4)))  float         f32x4;
typedef __attribute__((ext_vector_type(4)))  unsigned int  u32x4;
typedef __attribute__((ext_vector_type(4)))  int           i32x4v;

union Frag { u32x4 q[2]; v16bf v; };

__device__ __forceinline__ unsigned short f2bf(float f) {
  unsigned int u = __float_as_uint(f);
  unsigned int r = u + 0x7FFFu + ((u >> 16) & 1u);   // round-to-nearest-even
  return (unsigned short)(r >> 16);
}
__device__ __forceinline__ unsigned int pk2(float a, float b) {
  return (unsigned int)f2bf(a) | ((unsigned int)f2bf(b) << 16);
}
__device__ __forceinline__ float sigmoidf_fast(float x) {
  return 1.0f / (1.0f + __expf(-x));
}

// Async memory->LDS copy (CDNA5 GLOBAL_LOAD_ASYNC_TO_LDS_B128, ASYNCcnt),
// with a synchronous fallback so compilation never regresses.
#if __has_builtin(__builtin_amdgcn_global_load_async_to_lds_b128) && \
    __has_builtin(__builtin_amdgcn_s_wait_asynccnt)
#define USE_ASYNC_LDS 1
#else
#define USE_ASYNC_LDS 0
#endif

__device__ __forceinline__ void stage16(unsigned short* ldst,
                                        const unsigned short* gsrc) {
#if USE_ASYNC_LDS
  __builtin_amdgcn_global_load_async_to_lds_b128(
      (__attribute__((address_space(1))) i32x4v*)gsrc,
      (__attribute__((address_space(3))) i32x4v*)ldst, 0, 0);
#else
  *(u32x4*)ldst = *(const u32x4*)gsrc;
#endif
}
__device__ __forceinline__ void stage_wait() {
#if USE_ASYNC_LDS
  __builtin_amdgcn_s_wait_asynccnt(0);
#endif
}

// ---------------------------------------------------------------------------
// f32 -> bf16 conversion pass (8 elements/thread; all sizes divide 2048)
// ---------------------------------------------------------------------------
__global__ __launch_bounds__(256)
void cvt_bf16_kernel(const float* __restrict__ s, unsigned short* __restrict__ d) {
  const size_t i = (size_t)(blockIdx.x * 256u + threadIdx.x) * 8;
  f32x4 a = *(const f32x4*)(s + i);
  f32x4 b = *(const f32x4*)(s + i + 4);
  u32x4 p = {pk2(a[0], a[1]), pk2(a[2], a[3]), pk2(b[0], b[1]), pk2(b[2], b[3])};
  *(u32x4*)(d + i) = p;
}

// ---------------------------------------------------------------------------
// bf16 WMMA GEMM:  C[M,N] = A[M,K] @ W[N,K]^T   (bf16 operands, f32 accum)
// Block tile 128x128, K-tile 32, 256 threads = 8 waves (2 x 4), wave = 64x32.
// Double-buffered LDS; next tile staged via async LDS copies while the
// current tile's 8 WMMAs run.  EPI: 0 = plain, 1 = softplus(acc + bias[col])
// ---------------------------------------------------------------------------
template <int EPI>
__global__ __launch_bounds__(256)
void gemm_bf16_wmma(const unsigned short* __restrict__ A,
                    const unsigned short* __restrict__ W,
                    const float* __restrict__ bias, float* __restrict__ C,
                    int N, int K) {
  __shared__ __align__(16) unsigned short lA[2][128 * 32];
  __shared__ __align__(16) unsigned short lB[2][128 * 32];

  const int tid  = threadIdx.x;
  const int lane = tid & 31;
  const int wid  = tid >> 5;
  const int wm   = wid & 1;        // 0..1  -> 64 rows each
  const int wn   = wid >> 1;       // 0..3  -> 32 cols each
  const int bm   = blockIdx.y * 128;
  const int bn   = blockIdx.x * 128;

  v8f acc[4][2];
#pragma unroll
  for (int i = 0; i < 4; ++i)
#pragma unroll
    for (int j = 0; j < 2; ++j) acc[i][j] = (v8f)(0.0f);

  // staging: each thread copies 32B of A and 32B of W per K-tile
  const int lr = tid >> 1;          // 0..127
  const int lc = (tid & 1) * 16;    // 0 or 16
  const unsigned short* gA = A + (size_t)(bm + lr) * K + lc;
  const unsigned short* gW = W + (size_t)(bn + lr) * K + lc;
  const int soff = lr * 32 + lc;

  const int hg = lane >> 4;         // half-wave group (0/1)
  const int lm = lane & 15;

  // prologue: stage tile 0 into buffer 0
  stage16(&lA[0][soff], gA);
  stage16(&lA[0][soff + 8], gA + 8);
  stage16(&lB[0][soff], gW);
  stage16(&lB[0][soff + 8], gW + 8);

  const int steps = K >> 5;
  for (int it = 0; it < steps; ++it) {
    const int cur = it & 1;
    stage_wait();                 // this wave's async copies have landed
    __syncthreads();              // => all waves' copies of tile 'it' landed

    if (it + 1 < steps) {         // stage next tile into the other buffer
      const int kt = (it + 1) << 5;
      const int nb = cur ^ 1;
      stage16(&lA[nb][soff], gA + kt);
      stage16(&lA[nb][soff + 8], gA + kt + 8);
      stage16(&lB[nb][soff], gW + kt);
      stage16(&lB[nb][soff + 8], gW + kt + 8);
      if (it + 2 < steps) {       // warm L2 for the tile after that
        __builtin_prefetch(gA + kt + 32, 0, 1);
        __builtin_prefetch(gW + kt + 32, 0, 1);
      }
    }

    // --- fragments per CDNA5 ISA 16-bit layouts -----------------
    Frag af[4];
#pragma unroll
    for (int mt = 0; mt < 4; ++mt) {
      const unsigned short* p = &lA[cur][(wm * 64 + mt * 16 + lm) * 32];
      af[mt].q[0] = *(const u32x4*)(p + hg * 8);        // K = 0..7  (+8 hg=1)
      af[mt].q[1] = *(const u32x4*)(p + 16 + hg * 8);   // K = 16..23 (+8)
    }
    Frag bfrag[2];
#pragma unroll
    for (int nt = 0; nt < 2; ++nt) {
      const unsigned short* p = &lB[cur][(wn * 32 + nt * 16 + lm) * 32 + hg * 8];
      bfrag[nt].q[0] = *(const u32x4*)p;
      bfrag[nt].q[1] = *(const u32x4*)(p + 16);
    }
#pragma unroll
    for (int mt = 0; mt < 4; ++mt)
#pragma unroll
      for (int nt = 0; nt < 2; ++nt)
        acc[mt][nt] = __builtin_amdgcn_wmma_f32_16x16x32_bf16(
            false, af[mt].v, false, bfrag[nt].v, (short)0, acc[mt][nt],
            false, false);
  }

  // epilogue: VGPR r -> row r (lanes 0-15) / r+8 (lanes 16-31), col = lane&15
#pragma unroll
  for (int mt = 0; mt < 4; ++mt)
#pragma unroll
    for (int nt = 0; nt < 2; ++nt) {
      const int row0 = bm + wm * 64 + mt * 16 + hg * 8;
      const int col  = bn + wn * 32 + nt * 16 + lm;
      float bv = 0.0f;
      if (EPI == 1) bv = bias[col];
#pragma unroll
      for (int r = 0; r < 8; ++r) {
        float v = acc[mt][nt][r];
        if (EPI == 1) {            // softplus(v + dt_b)
          v += bv;
          v = (v > 20.0f) ? v : log1pf(__expf(v));
        }
        C[(size_t)(row0 + r) * N + col] = v;
      }
    }
}

// ---------------------------------------------------------------------------
// Causal depthwise conv1d (K=4) + bias + SiLU.  x_p = xz[..., :Din]
// Writes f32 (for bcproj/scan) and bf16 (feeds the dt GEMM directly).
// ---------------------------------------------------------------------------
__global__ __launch_bounds__(256)
void conv_silu_kernel(const float* __restrict__ xz, const float* __restrict__ cw,
                      const float* __restrict__ cb, float* __restrict__ xs,
                      unsigned short* __restrict__ xsb) {
  const unsigned idx = blockIdx.x * 256u + threadIdx.x;  // B*L*Din = 2^23 exact
  const int d = idx & 2047;
  const int l = (idx >> 11) & 2047;
  const int b = idx >> 22;
  const float* base = xz + ((size_t)(b * 2048 + l) * 4096) + d;
  float acc = cb[d];
#pragma unroll
  for (int j = 0; j < 4; ++j) {
    const int ls = l - 3 + j;
    if (ls >= 0) acc = fmaf(cw[d * 4 + j], base[(ptrdiff_t)(j - 3) * 4096], acc);
  }
  const float s = acc * sigmoidf_fast(acc);
  xs[idx]  = s;
  xsb[idx] = f2bf(s);
}

// ---------------------------------------------------------------------------
// B/C projections: one wave per row; lanes 0-15 -> B_proj, 16-31 -> C_proj.
// ---------------------------------------------------------------------------
__global__ __launch_bounds__(256)
void bcproj_kernel(const float* __restrict__ xs, const float* __restrict__ Bw,
                   const float* __restrict__ Cw, float* __restrict__ Bp,
                   float* __restrict__ Cp) {
  const int row  = blockIdx.x * 8 + (threadIdx.x >> 5);
  const int lane = threadIdx.x & 31;
  const float* w = (lane < 16) ? (Bw + (size_t)lane * 2048)
                               : (Cw + (size_t)(lane - 16) * 2048);
  const f32x4* xr = (const f32x4*)(xs + (size_t)row * 2048);
  const f32x4* wr = (const f32x4*)w;
  float acc = 0.0f;
  for (int k = 0; k < 512; ++k) {
    f32x4 a = xr[k], b = wr[k];
    acc = fmaf(a[0], b[0], acc);
    acc = fmaf(a[1], b[1], acc);
    acc = fmaf(a[2], b[2], acc);
    acc = fmaf(a[3], b[3], acc);
  }
  if (lane < 16) Bp[(size_t)row * 16 + lane] = acc;
  else           Cp[(size_t)row * 16 + (lane - 16)] = acc;
}

// ---------------------------------------------------------------------------
// Selective scan: one lane per (b, d, n) state; 16-lane butterfly reduction;
// fused D-skip and SiLU(z) gate; emits bf16 directly for the out GEMM.
// ---------------------------------------------------------------------------
__global__ __launch_bounds__(256)
void scan_kernel(const float* __restrict__ dt, const float* __restrict__ xs,
                 const float* __restrict__ Bp, const float* __restrict__ Cp,
                 const float* __restrict__ A_log, const float* __restrict__ Dp,
                 const float* __restrict__ xz, unsigned short* __restrict__ ygb) {
  const int idx = blockIdx.x * 256 + threadIdx.x;  // 65536 = B*Din*16
  const int n  = idx & 15;
  const int ch = idx >> 4;
  const int d  = ch & 2047;
  const int b  = ch >> 11;
  const float And = -__expf(A_log[d * 16 + n]);
  const float Dd  = Dp[d];
  float h = 0.0f;
  for (int t = 0; t < 2048; ++t) {
    const size_t r = (size_t)(b * 2048 + t);
    const float dtv = dt[r * 2048 + d];
    const float xv  = xs[r * 2048 + d];
    const float Bv  = Bp[r * 16 + n];
    const float Cv  = Cp[r * 16 + n];
    const float a = __expf(dtv * And);
    h = fmaf(a, h, dtv * Bv * xv);
    float contrib = h * Cv;
    contrib += __shfl_xor(contrib, 1, 16);
    contrib += __shfl_xor(contrib, 2, 16);
    contrib += __shfl_xor(contrib, 4, 16);
    contrib += __shfl_xor(contrib, 8, 16);
    if (n == 0) {
      const float yt = contrib + Dd * xv;
      const float z  = xz[r * 4096 + 2048 + d];
      ygb[r * 2048 + d] = f2bf(yt * z * sigmoidf_fast(z));
    }
  }
}

// ---------------------------------------------------------------------------
// Launch
// ---------------------------------------------------------------------------
extern "C" void kernel_launch(void* const* d_in, const int* in_sizes, int n_in,
                              void* d_out, int out_size, void* d_ws, size_t ws_size,
                              hipStream_t stream) {
  (void)in_sizes; (void)n_in; (void)out_size; (void)ws_size;
  const float* x         = (const float*)d_in[0];
  const float* in_proj_w = (const float*)d_in[1];
  const float* conv_w    = (const float*)d_in[2];
  const float* conv_b    = (const float*)d_in[3];
  const float* dt_w      = (const float*)d_in[4];
  const float* dt_b      = (const float*)d_in[5];
  const float* B_proj_w  = (const float*)d_in[6];
  const float* C_proj_w  = (const float*)d_in[7];
  const float* A_log     = (const float*)d_in[8];
  const float* D_param   = (const float*)d_in[9];
  const float* out_proj_w= (const float*)d_in[10];
  float* out = (float*)d_out;

  // f32 workspace: xz 16.78M | xs 8.39M | dt 8.39M | Bp/Cp 64K each
  float* ws  = (float*)d_ws;
  float* xz  = ws;
  float* xs  = ws + 16777216;
  float* dtv = ws + 25165824;
  float* Bp  = ws + 33554432;
  float* Cp  = ws + 33619968;
  // bf16 workspace (ushort), after the f32 region
  unsigned short* us  = (unsigned short*)(ws + 33685504);
  unsigned short* xb  = us;              // x            4,194,304
  unsigned short* wib = us + 4194304;    // in_proj_w    4,194,304
  unsigned short* wdb = us + 8388608;    // dt_w         4,194,304
  unsigned short* wob = us + 12582912;   // out_proj_w   2,097,152
  unsigned short* xsb = us + 14680064;   // xs           8,388,608
  unsigned short* ygb = us + 23068672;   // gated y      8,388,608

  const dim3 blk(256);
  // 0) one-time bf16 conversions (removes pack work + halves GEMM traffic)
  cvt_bf16_kernel<<<2048, blk, 0, stream>>>(x,          xb);
  cvt_bf16_kernel<<<2048, blk, 0, stream>>>(in_proj_w,  wib);
  cvt_bf16_kernel<<<2048, blk, 0, stream>>>(dt_w,       wdb);
  cvt_bf16_kernel<<<1024, blk, 0, stream>>>(out_proj_w, wob);
  // 1) in_proj: (4096 x 4096 x 1024)
  gemm_bf16_wmma<0><<<dim3(4096 / 128, 4096 / 128), blk, 0, stream>>>(
      xb, wib, nullptr, xz, 4096, 1024);
  // 2) causal depthwise conv + SiLU (f32 + bf16 outputs)
  conv_silu_kernel<<<32768, blk, 0, stream>>>(xz, conv_w, conv_b, xs, xsb);
  // 3) dt = softplus(xs @ dt_w^T + dt_b): (4096 x 2048 x 2048)
  gemm_bf16_wmma<1><<<dim3(2048 / 128, 4096 / 128), blk, 0, stream>>>(
      xsb, wdb, dt_b, dtv, 2048, 2048);
  // 4) B/C projections
  bcproj_kernel<<<512, blk, 0, stream>>>(xs, B_proj_w, C_proj_w, Bp, Cp);
  // 5) selective scan + D-skip + SiLU(z) gate -> bf16
  scan_kernel<<<256, blk, 0, stream>>>(dtv, xs, Bp, Cp, A_log, D_param, xz, ygb);
  // 6) out_proj: (4096 x 1024 x 2048)
  gemm_bf16_wmma<0><<<dim3(1024 / 128, 4096 / 128), blk, 0, stream>>>(
      ygb, wob, nullptr, out, 1024, 2048);
}
